// FiLMKTokenKeyModel_66760971649634
// MI455X (gfx1250) — matrix operand order repaired and checked by reading, not compiled, not run on hardware
//
#include <hip/hip_runtime.h>
#include <hip/hip_bf16.h>
#include <stdint.h>

// ---------------- problem constants ----------------
#define BB    4
#define TT    2048
#define DD    1024
#define DB    256
#define DK    128
#define KV    48
#define DIN   1408          // DD + DB + DK
#define NTOK  (BB*TT)       // 8192
#define NOUT2 2048          // 2*DD
#define RWIN  8
#define NWIN  17            // 2R+1
#define CCH   18            // 1 + NWIN
#define KT32  (DIN/32)      // 44 k-steps of 32

typedef __attribute__((ext_vector_type(16))) __bf16 v16bf;
typedef __attribute__((ext_vector_type(8)))  __bf16 v8bf;
typedef __attribute__((ext_vector_type(8)))  float  v8f;

// gfx1250 async Global->LDS copy path (ASYNCcnt), guarded so a toolchain
// without the builtin falls back to sync load + ds_store staging.
#if defined(__AMDGCN__) && __has_builtin(__builtin_amdgcn_global_load_async_to_lds_b128)
#define HAVE_ASYNC 1
#else
#define HAVE_ASYNC 0
#endif

#if HAVE_ASYNC
typedef int v4i_vs __attribute__((vector_size(16)));
typedef __attribute__((address_space(1))) v4i_vs* gptr_v4i;   // global src
typedef __attribute__((address_space(3))) v4i_vs* lptr_v4i;   // LDS dst
#endif

static __device__ inline unsigned short f2bf(float f) {
    union { float f; unsigned u; } v; v.f = f;
    unsigned u = v.u;
    unsigned r = u + 0x7FFFu + ((u >> 16) & 1u);   // RNE
    return (unsigned short)(r >> 16);
}

// ---------------------------------------------------------------------------
// Kernel 1: film_w [1408,2048] f32  ->  film_wT [2048,1408] bf16
// ---------------------------------------------------------------------------
__global__ __launch_bounds__(256) void transpose_w(const float* __restrict__ fw,
                                                   unsigned short* __restrict__ fwT) {
    __shared__ float tile[32][33];
    const int nBase = blockIdx.x * 32;   // over N = 2048
    const int kBase = blockIdx.y * 32;   // over K = 1408
    const int tx = threadIdx.x;          // 0..31
    const int ty = threadIdx.y;          // 0..7
#pragma unroll
    for (int i = 0; i < 4; ++i) {
        int r = ty + i * 8;
        tile[r][tx] = fw[(size_t)(kBase + r) * NOUT2 + nBase + tx];
    }
    __syncthreads();
#pragma unroll
    for (int i = 0; i < 4; ++i) {
        int r = ty + i * 8;
        fwT[(size_t)(nBase + r) * DIN + kBase + tx] = f2bf(tile[tx][r]);
    }
}

// ---------------------------------------------------------------------------
// Kernel 2: per-token (one wave32 per token):
//   bl[tok]  = h . bound_w + bound_b
//   ek[tok,:] = softmax(h @ key_w + key_b) @ key_emb
// ---------------------------------------------------------------------------
__global__ __launch_bounds__(256) void tok_stage1(const float* __restrict__ h,
                                                  const float* __restrict__ bound_w,
                                                  const float* __restrict__ bound_b,
                                                  const float* __restrict__ key_w,
                                                  const float* __restrict__ key_b,
                                                  const float* __restrict__ key_emb,
                                                  float* __restrict__ bl,
                                                  float* __restrict__ ek) {
    const int wave = threadIdx.x >> 5;
    const int lane = threadIdx.x & 31;
    const int tok  = blockIdx.x * 8 + wave;
    const float* hrow = h + (size_t)tok * DD;

    float hbuf[32];
#pragma unroll
    for (int i = 0; i < 8; ++i) {
        float4 v = *(const float4*)(hrow + lane * 32 + i * 4);
        hbuf[i*4+0] = v.x; hbuf[i*4+1] = v.y; hbuf[i*4+2] = v.z; hbuf[i*4+3] = v.w;
    }

    float bacc = 0.f;
    float kacc[KV];
#pragma unroll
    for (int k = 0; k < KV; ++k) kacc[k] = 0.f;

#pragma unroll 4
    for (int i = 0; i < 32; ++i) {
        const int d = lane * 32 + i;
        const float hv = hbuf[i];
        bacc += hv * bound_w[d];
        const float* kw = key_w + (size_t)d * KV;
#pragma unroll
        for (int k4 = 0; k4 < KV / 4; ++k4) {
            float4 w = *(const float4*)(kw + k4 * 4);
            kacc[k4*4+0] += hv * w.x;
            kacc[k4*4+1] += hv * w.y;
            kacc[k4*4+2] += hv * w.z;
            kacc[k4*4+3] += hv * w.w;
        }
    }

    // butterfly reduce across the wave (all lanes end with full sums)
    for (int off = 16; off > 0; off >>= 1) {
        bacc += __shfl_xor(bacc, off, 32);
#pragma unroll
        for (int k = 0; k < KV; ++k) kacc[k] += __shfl_xor(kacc[k], off, 32);
    }

    if (lane == 0) bl[tok] = bacc + bound_b[0];

    // softmax over 48 (redundant per-lane, tiny)
    float kl[KV], m = -1e30f;
#pragma unroll
    for (int k = 0; k < KV; ++k) { kl[k] = kacc[k] + key_b[k]; m = fmaxf(m, kl[k]); }
    float s = 0.f;
#pragma unroll
    for (int k = 0; k < KV; ++k) { kl[k] = __expf(kl[k] - m); s += kl[k]; }
    const float inv = 1.f / s;

    // ek = soft @ key_emb  (each lane: 4 output features, coalesced)
#pragma unroll
    for (int el = 0; el < 4; ++el) {
        const int e = el * 32 + lane;
        float acc = 0.f;
#pragma unroll
        for (int k = 0; k < KV; ++k) acc += kl[k] * key_emb[k * DK + e];
        ek[(size_t)tok * DK + e] = acc * inv;
    }
}

// ---------------------------------------------------------------------------
// Kernel 3: per-token (one wave32 per token):
//   b_soft -> eb ; layernorm(concat[h,eb,ek]) -> bf16 film_in ; mu/rsig of h
// ---------------------------------------------------------------------------
__global__ __launch_bounds__(256) void tok_stage2(const float* __restrict__ h,
                                                  const float* __restrict__ conv_w,
                                                  const float* __restrict__ conv_b,
                                                  const float* __restrict__ e0,
                                                  const float* __restrict__ e1,
                                                  const float* __restrict__ bl,
                                                  const float* __restrict__ ek,
                                                  const float* __restrict__ ln_in_g,
                                                  const float* __restrict__ ln_in_b,
                                                  unsigned short* __restrict__ fin,
                                                  float* __restrict__ muh,
                                                  float* __restrict__ rsigh) {
    const int wave = threadIdx.x >> 5;
    const int lane = threadIdx.x & 31;
    const int tok  = blockIdx.x * 8 + wave;
    const int b = tok >> 11;
    const int t = tok & (TT - 1);

    // conv9 (zero pad) + sigmoid
    float sm = 0.f;
#pragma unroll
    for (int j = 0; j < 9; ++j) {
        int tt = t + j - 4;
        if (tt >= 0 && tt < TT) sm += conv_w[j] * bl[(b << 11) + tt];
    }
    const float bs = 1.f / (1.f + __expf(-(sm + conv_b[0])));

    float xv[KT32];
    float s1 = 0.f, s2 = 0.f, sh1 = 0.f, sh2 = 0.f;
#pragma unroll
    for (int j = 0; j < KT32; ++j) {
        const int i = lane + (j << 5);
        float x;
        if (j < 32)       x = h[(size_t)tok * DD + i];
        else if (j < 40)  { int ii = i - DD; x = bs * e1[ii] + (1.f - bs) * e0[ii]; }
        else              x = ek[(size_t)tok * DK + (i - (DD + DB))];
        xv[j] = x; s1 += x; s2 += x * x;
        if (j < 32) { sh1 += x; sh2 += x * x; }
    }
    for (int off = 16; off > 0; off >>= 1) {
        s1  += __shfl_xor(s1,  off, 32);
        s2  += __shfl_xor(s2,  off, 32);
        sh1 += __shfl_xor(sh1, off, 32);
        sh2 += __shfl_xor(sh2, off, 32);
    }
    const float mu  = s1 * (1.f / DIN);
    const float rs  = rsqrtf(s2 * (1.f / DIN) - mu * mu + 1e-5f);
    const float muv = sh1 * (1.f / DD);
    const float rsh = rsqrtf(sh2 * (1.f / DD) - muv * muv + 1e-5f);

#pragma unroll
    for (int j = 0; j < KT32; ++j) {
        const int i = lane + (j << 5);
        float y = (xv[j] - mu) * rs * ln_in_g[i] + ln_in_b[i];
        fin[(size_t)tok * DIN + i] = f2bf(y);
    }
    if (lane == 0) { muh[tok] = muv; rsigh[tok] = rsh; }
}

// ---------------------------------------------------------------------------
// Kernel 4: WMMA GEMM  gb = film_in(bf16) @ film_wT(bf16) + film_b, fused
//   epilogue z = lnorm(h)*(1+gamma)+beta -> out[:, :, 0, :]
//   block: 256 thr (8 waves). tile: M64 x (N64 gamma + N64 beta). K = 44*32.
//   Double-buffered LDS; async Global->LDS (ASYNCcnt) when available.
// ---------------------------------------------------------------------------
#define LDKS 40   // LDS row stride in bf16 (32 data + 8 pad), 80B, 16B-aligned

// 16B copy Global -> LDS (async on gfx1250, sync fallback otherwise)
static __device__ inline void cp16(const unsigned short* g, __bf16* l) {
#if HAVE_ASYNC
    __builtin_amdgcn_global_load_async_to_lds_b128(
        (gptr_v4i)(v4i_vs*)g, (lptr_v4i)(v4i_vs*)l, 0, 0);
#else
    *(uint4*)l = *(const uint4*)g;
#endif
}
static __device__ inline void wait_async_copies() {
#if HAVE_ASYNC
#if __has_builtin(__builtin_amdgcn_s_wait_asynccnt)
    __builtin_amdgcn_s_wait_asynccnt(0);
#else
    asm volatile("s_wait_asynccnt 0x0" ::: "memory");
#endif
#endif
}

static __device__ inline v16bf load_fragA(const __bf16* base, int row0, int lane) {
    // A 16x32 bf16: lanes 0-15 rows, K {0-7,16-23}; lanes 16-31 K {8-15,24-31}
    const __bf16* p = base + (size_t)(row0 + (lane & 15)) * LDKS + ((lane >> 4) << 3);
    v8bf lo = *(const v8bf*)(p);
    v8bf hi = *(const v8bf*)(p + 16);
    return __builtin_shufflevector(lo, hi, 0,1,2,3,4,5,6,7,8,9,10,11,12,13,14,15);
}
static __device__ inline v16bf load_fragB(const __bf16* base, int row0, int lane) {
    // B 32x16 bf16 (col-major rows in fwT): lanes 0-15 K {0-7,16-23}; lanes 16-31 K {8-15,24-31}
    const __bf16* p = base + (size_t)(row0 + (lane & 15)) * LDKS + ((lane >> 4) << 3);
    v8bf lo = *(const v8bf*)(p);
    v8bf hi = *(const v8bf*)(p + 16);
    return __builtin_shufflevector(lo, hi, 0,1,2,3,4,5,6,7,8,9,10,11,12,13,14,15);
}
static __device__ inline v8f wmma_bf16(v16bf a, v16bf b, v8f c) {
    return __builtin_amdgcn_wmma_f32_16x16x32_bf16(false, a, false, b, (short)0, c,
                                                   false, false);
}

__global__ __launch_bounds__(256) void film_gemm(const unsigned short* __restrict__ fin,
                                                 const unsigned short* __restrict__ fwT,
                                                 const float* __restrict__ film_b,
                                                 const float* __restrict__ h,
                                                 const float* __restrict__ muh,
                                                 const float* __restrict__ rsigh,
                                                 const float* __restrict__ ln_h_g,
                                                 const float* __restrict__ ln_h_b,
                                                 float* __restrict__ out) {
    __shared__ __align__(16) __bf16 As[2][64 * LDKS];
    __shared__ __align__(16) __bf16 Bg[2][64 * LDKS];
    __shared__ __align__(16) __bf16 Bb[2][64 * LDKS];

    const int tid  = threadIdx.x;
    const int wave = tid >> 5;
    const int lane = tid & 31;
    const int m0   = blockIdx.x * 64;       // token tile
    const int n0   = blockIdx.y * 64;       // gamma-column tile (beta = +1024)
    const int r    = tid >> 2;              // 0..63 loader row
    const int q    = tid & 3;               // 16B chunk within 64B row
    const int miA  = (wave & 1) * 32;       // first m-subtile row offset in As
    const int ni   = (wave >> 1) * 16;      // n-subtile row offset in Bg/Bb

    const unsigned short* gA = fin + (size_t)(m0 + r) * DIN + q * 8;
    const unsigned short* gG = fwT + (size_t)(n0 + r) * DIN + q * 8;
    const unsigned short* gB = fwT + (size_t)(DD + n0 + r) * DIN + q * 8;
    const int lofs = r * LDKS + q * 8;

    v8f ag0 = {}, ag1 = {}, ab0 = {}, ab1 = {};

    // prologue: fill buffer 0 with k-tile 0
    cp16(gA, &As[0][lofs]);
    cp16(gG, &Bg[0][lofs]);
    cp16(gB, &Bb[0][lofs]);
    wait_async_copies();
    __syncthreads();

    for (int kt = 0; kt < KT32; ++kt) {
        const int cur = kt & 1;
        if (kt + 1 < KT32) {
            // overlap: async-stage next k-tile into the other buffer while
            // this iteration's WMMAs consume the current one
            const size_t ko = (size_t)(kt + 1) * 32;
            cp16(gA + ko, &As[cur ^ 1][lofs]);
            cp16(gG + ko, &Bg[cur ^ 1][lofs]);
            cp16(gB + ko, &Bb[cur ^ 1][lofs]);
        }
#if !HAVE_ASYNC
        if (kt + 2 < KT32)  // near-cache hint for the tile after next
            __builtin_prefetch(gA + (size_t)(kt + 2) * 32, 0, 3);
#endif

        v16bf a0 = load_fragA(&As[cur][0], miA,      lane);
        v16bf a1 = load_fragA(&As[cur][0], miA + 16, lane);
        v16bf bg = load_fragB(&Bg[cur][0], ni, lane);
        v16bf bbf= load_fragB(&Bb[cur][0], ni, lane);
        ag0 = wmma_bf16(a0, bg,  ag0);
        ag1 = wmma_bf16(a1, bg,  ag1);
        ab0 = wmma_bf16(a0, bbf, ab0);
        ab1 = wmma_bf16(a1, bbf, ab1);

        wait_async_copies();   // next buffer fully in LDS (this wave's copies)
        __syncthreads();       // all waves: reads of cur done, nxt visible
    }

    // fused epilogue: z = ((h-mu)*rsig*g + b) * (1+gamma) + beta  -> out[.,.,0,.]
    const int ncol = n0 + ni + (lane & 15);
    const float fbg = film_b[ncol];
    const float fbb = film_b[DD + ncol];
    const float gg  = ln_h_g[ncol];
    const float gb  = ln_h_b[ncol];
    const int mhi   = (lane >> 4) * 8;

#pragma unroll
    for (int half = 0; half < 2; ++half) {
        const v8f AG = half ? ag1 : ag0;
        const v8f AB = half ? ab1 : ab0;
        const int mbase = m0 + miA + half * 16 + mhi;
#pragma unroll
        for (int j = 0; j < 8; ++j) {
            const int m = mbase + j;
            const float gamma = AG[j] + fbg;
            const float beta  = AB[j] + fbb;
            const float hv = h[(size_t)m * DD + ncol];
            const float lnh = (hv - muh[m]) * rsigh[m] * gg + gb;
            out[(size_t)m * (CCH * DD) + ncol] = lnh * (1.f + gamma) + beta;
        }
    }
}

// ---------------------------------------------------------------------------
// Kernel 5: sliding windows (edge-clamped) -> out[:, :, 1:18, :], float4
// ---------------------------------------------------------------------------
__global__ __launch_bounds__(256) void windows(const float* __restrict__ h,
                                               float* __restrict__ out) {
    const size_t idx = (size_t)blockIdx.x * 256 + threadIdx.x;   // float4 units
    const size_t total4 = (size_t)NTOK * NWIN * (DD / 4);
    if (idx >= total4) return;
    const int d4 = (int)(idx & (DD / 4 - 1));
    size_t rest = idx >> 8;                 // / (DD/4)
    const int w = (int)(rest % NWIN);
    rest /= NWIN;
    const int t = (int)(rest & (TT - 1));
    const int b = (int)(rest >> 11);
    int st = t + w - RWIN;
    st = st < 0 ? 0 : (st > TT - 1 ? TT - 1 : st);
    float4 v = *(const float4*)(h + (((size_t)b * TT + st) << 10) + (d4 << 2));
    *(float4*)(out + ((((size_t)b * TT + t) * CCH + 1 + w) << 10) + (d4 << 2)) = v;
}

// ---------------------------------------------------------------------------
extern "C" void kernel_launch(void* const* d_in, const int* in_sizes, int n_in,
                              void* d_out, int out_size, void* d_ws, size_t ws_size,
                              hipStream_t stream) {
    const float* h       = (const float*)d_in[0];
    const float* bound_w = (const float*)d_in[1];
    const float* bound_b = (const float*)d_in[2];
    const float* conv_w  = (const float*)d_in[3];
    const float* conv_b  = (const float*)d_in[4];
    const float* e0      = (const float*)d_in[5];
    const float* e1      = (const float*)d_in[6];
    const float* key_emb = (const float*)d_in[7];
    const float* key_w   = (const float*)d_in[8];
    const float* key_b   = (const float*)d_in[9];
    const float* ln_in_g = (const float*)d_in[10];
    const float* ln_in_b = (const float*)d_in[11];
    const float* film_w  = (const float*)d_in[12];
    const float* film_b  = (const float*)d_in[13];
    const float* ln_h_g  = (const float*)d_in[14];
    const float* ln_h_b  = (const float*)d_in[15];
    float* out = (float*)d_out;

    char* ws = (char*)d_ws;
    float*          bl    = (float*)(ws);                                  // 32 KB
    float*          muh   = (float*)(ws + 32768);                          // 32 KB
    float*          rsigh = (float*)(ws + 65536);                          // 32 KB
    float*          ek    = (float*)(ws + 98304);                          // 4 MB
    unsigned short* fin   = (unsigned short*)(ws + 98304 + 4194304);       // 23.1 MB
    unsigned short* fwT   = (unsigned short*)(ws + 98304 + 4194304 + 23068672); // 5.77 MB

    // 1. weight transpose/convert (independent of token kernels)
    transpose_w<<<dim3(NOUT2 / 32, DIN / 32), dim3(32, 8), 0, stream>>>(film_w, fwT);
    // 2. boundary logits + key attention embedding
    tok_stage1<<<NTOK / 8, 256, 0, stream>>>(h, bound_w, bound_b, key_w, key_b,
                                             key_emb, bl, ek);
    // 3. conv/sigmoid + film_in layernorm (bf16) + h moments
    tok_stage2<<<NTOK / 8, 256, 0, stream>>>(h, conv_w, conv_b, e0, e1, bl, ek,
                                             ln_in_g, ln_in_b, fin, muh, rsigh);
    // 4. WMMA GEMM + fused FiLM epilogue -> out channel 0
    film_gemm<<<dim3(NTOK / 64, DD / 64), 256, 0, stream>>>(
        fin, fwT, film_b, h, muh, rsigh, ln_h_g, ln_h_b, out);
    // 5. sliding windows -> out channels 1..17
    const size_t total4 = (size_t)NTOK * NWIN * (DD / 4);
    windows<<<(unsigned)((total4 + 255) / 256), 256, 0, stream>>>(h, out);
}
